// QConv1d_12128987644269
// MI455X (gfx1250) — matrix operand order repaired and checked by reading, not compiled, not run on hardware
//
#include <hip/hip_runtime.h>

// ---- CDNA5 (gfx1250) wave32 WMMA types ----
typedef __attribute__((ext_vector_type(16))) __bf16 v16bf;
typedef __attribute__((ext_vector_type(8)))  __bf16 v8bf;
typedef __attribute__((ext_vector_type(8)))  float  v8f;

#define L_IN     16384
#define N_BATCH  8
#define C_IN     64
#define C_OUT    64
#define KW       9
#define PADW     4
#define QSCALE   0.125f

#define TILE_P   128                 // output positions per workgroup
#define HALO     (2 * PADW)          // 8
#define XS_P     (TILE_P + HALO)     // 136 staged positions
#define XS_STRIDE 72                 // padded channel row stride (16B-aligned rows, bank spread)
#define NCHUNK   18                  // K=576 -> 18 chunks of 32 (9 taps x 2 channel halves)

static __device__ __forceinline__ unsigned short f32_to_bf16_rne(float f) {
    union { float f; unsigned int u; } v; v.f = f;
    unsigned int u = v.u;
    unsigned int r = u + 0x7FFFu + ((u >> 16) & 1u);   // round-to-nearest-even
    return (unsigned short)(r >> 16);
}

// Prepack weight (O,I,KW) f32 -> bf16 in the exact 16-bit A-matrix (16x32) VGPR
// layout of V_WMMA_F32_16X16X32_BF16:  element i of lane L holds
//   A[m = L&15][k = (i>>3)*16 + (L>>4)*8 + (i&7)]
// where logical k within chunk kc maps to channel c0 + k at tap t = kc>>1,
// c0 = (kc&1)*32.  One block per (o-tile, chunk); 512 threads = 32 lanes x 16 elems.
__global__ __launch_bounds__(512) void prepack_weight_kernel(
    const float* __restrict__ w, unsigned short* __restrict__ pk) {
    const int ot   = blockIdx.x / NCHUNK;
    const int kc   = blockIdx.x % NCHUNK;
    const int lane = threadIdx.x >> 4;
    const int i    = threadIdx.x & 15;
    const int hi   = lane >> 4;
    const int m    = lane & 15;
    const int t    = kc >> 1;
    const int c0   = (kc & 1) << 5;
    const int kA   = ((i >> 3) << 4) + (hi << 3) + (i & 7);
    const int c    = c0 + kA;
    const int o    = (ot << 4) + m;
    const float val = w[((size_t)o * C_IN + c) * KW + t];
    pk[(((size_t)ot * NCHUNK + kc) * 32 + lane) * 16 + i] = f32_to_bf16_rne(val);
}

// Implicit-GEMM conv1d: one WG = 1 batch x 128 output positions, 4 waves,
// wave w computes o-channels [16w, 16w+16).  x tile staged to LDS as bf16
// transposed [p_local][c] so B fragments are two 16B LDS reads per lane.
__global__ __launch_bounds__(128) void qconv1d_wmma_kernel(
    const float* __restrict__ x, const unsigned short* __restrict__ pk,
    const float* __restrict__ bias, float* __restrict__ out) {
    __shared__ unsigned short xs[XS_P * XS_STRIDE];   // 19,584 B

    const int n    = blockIdx.y;
    const int p0   = blockIdx.x * TILE_P;
    const int tid  = threadIdx.x;
    const int wid  = tid >> 5;
    const int lane = tid & 31;

    // ---- stage x tile (f32 -> bf16, transpose to [p_local][c]) ----
    const float* xn = x + (size_t)n * C_IN * L_IN;
    for (int c = wid; c < C_IN; c += 4) {
        const float* xc = xn + (size_t)c * L_IN;
        for (int pl = lane; pl < XS_P; pl += 32) {          // coalesced along L
            const int l = p0 - PADW + pl;
            const float v = (l >= 0 && l < L_IN) ? xc[l] : 0.0f;
            xs[pl * XS_STRIDE + c] = f32_to_bf16_rne(v);
        }
    }
    __syncthreads();

    const int hi   = lane >> 4;
    const int ncol = lane & 15;
    const int o0   = wid << 4;

    v8f acc[8];
#pragma unroll
    for (int pt = 0; pt < 8; ++pt) acc[pt] = (v8f){};

    const unsigned short* pkw =
        pk + (size_t)wid * NCHUNK * 32 * 16 + (size_t)lane * 16;

    for (int kc = 0; kc < NCHUNK; ++kc) {
        const int t  = kc >> 1;
        const int c0 = (kc & 1) << 5;
        // A fragment: 16 bf16/lane, coalesced 32B vector load (prepacked layout)
        const v16bf A = *(const v16bf*)(pkw + (size_t)kc * 32 * 16);
#pragma unroll
        for (int pt = 0; pt < 8; ++pt) {
            // B element i of lane L holds B[k = (i>>3)*16 + 8*hi + (i&7)][n = L&15]
            //   = x[c0 + k][p0 + 16*pt + ncol + t - PADW]
            const int pl = (pt << 4) + ncol + t;            // LDS already holds p0-PADW..
            const unsigned short* bp = &xs[pl * XS_STRIDE + c0 + (hi << 3)];
            union { v16bf v; v8bf h[2]; } B;
            B.h[0] = *(const v8bf*)(bp);                    // 16B-aligned ds_load
            B.h[1] = *(const v8bf*)(bp + 16);
            acc[pt] = __builtin_amdgcn_wmma_f32_16x16x32_bf16(
                /*neg_a=*/false, A, /*neg_b=*/false, B.v,
                /*c_mod=*/(short)0, acc[pt],
                /*reuse_a=*/false, /*reuse_b=*/false);
        }
    }

    // ---- epilogue: D layout m = r + 8*hi, n = ncol; apply scale + bias ----
    float bv[8];
#pragma unroll
    for (int r = 0; r < 8; ++r) bv[r] = bias[o0 + (hi << 3) + r];

#pragma unroll
    for (int pt = 0; pt < 8; ++pt) {
#pragma unroll
        for (int r = 0; r < 8; ++r) {
            const int o = o0 + (hi << 3) + r;
            const size_t oi = ((size_t)n * C_OUT + o) * L_IN
                              + (size_t)p0 + (pt << 4) + ncol;
            out[oi] = acc[pt][r] * QSCALE + bv[r];          // coalesced across lanes 0-15 / 16-31
        }
    }
}

extern "C" void kernel_launch(void* const* d_in, const int* in_sizes, int n_in,
                              void* d_out, int out_size, void* d_ws, size_t ws_size,
                              hipStream_t stream) {
    (void)in_sizes; (void)n_in; (void)out_size; (void)ws_size;
    const float* x    = (const float*)d_in[0];   // (8, 64, 16384)
    const float* w    = (const float*)d_in[1];   // (64, 64, 9)
    const float* bias = (const float*)d_in[2];   // (64,)
    float*       out  = (float*)d_out;           // (8, 64, 16384)
    unsigned short* pk = (unsigned short*)d_ws;  // 4*18*32*16 bf16 = 73,728 B

    prepack_weight_kernel<<<dim3(4 * NCHUNK), dim3(512), 0, stream>>>(w, pk);
    qconv1d_wmma_kernel<<<dim3(L_IN / TILE_P, N_BATCH), dim3(128), 0, stream>>>(
        x, pk, bias, out);
}
